// BallChallengeModel_46849503264970
// MI455X (gfx1250) — matrix operand order repaired and checked by reading, not compiled, not run on hardware
//
#include <hip/hip_runtime.h>

typedef __attribute__((ext_vector_type(16))) _Float16 v16h;
typedef __attribute__((ext_vector_type(8)))  _Float16 v8h;
typedef __attribute__((ext_vector_type(8)))  float    v8f;
typedef unsigned int u32;
typedef __attribute__((ext_vector_type(4))) u32 u32x4;
typedef __attribute__((ext_vector_type(8))) int i32x8;
typedef __attribute__((ext_vector_type(4))) int i32x4;

#define SEQ     8192
#define LFINAL  7949      // 8192 - 243 (total receptive field of 5 convs)
#define KPAD_L  7968      // LFINAL padded to multiple of 32 (249 chunks)
#define TILE    512       // final-output positions per conv workgroup
#define NTILES  16        // ceil(7949/512)

// ---- workspace byte offsets ----
#define OFF_W     0          // padded f16 conv weights: [16][32] + 4x[16][256] = 16896 f16
#define OFF_BIAS  33792      // 5*16 f32 quantized conv biases (contiguous after weights)
#define OFF_BL    34112      // 256 f32 quantized linear bias
#define OFF_WLT   36864      // [KPAD_L][256] f16 transposed quantized linear weight
#define OFF_H5    4194304    // [512][KPAD_L] f16 post-qhardtanh flattened features

// ---- LDS layout (single block, hand-carved so TDM descriptors have exact offsets) ----
#define S_X_OFF     0        // 3*784 f16            = 4704
#define S_H1_OFF    4704     // 16*752 f16           = 24064
#define S_H2_OFF    28768    // 8*736 f16            = 11776
#define S_H3_OFF    40544    // 4*704 f16            = 5632
#define S_H4_OFF    46176    // 2*640 f16            = 2560
#define S_W_OFF     48736    // 16896 f16            = 33792
#define S_BIAS_OFF  82528    // 80 f32               = 320   (contiguous after weights)
#define S_XSTG_OFF  82848    // 3*768 f32 raw x tile = 9216
#define S_TOTAL     92064

// Q8.8 quantize: round-half-even (rintf == jnp.round), clamp to int16 range.
__device__ __forceinline__ float quantize_q88(float x) {
  float q = rintf(x * 256.0f) * 0.00390625f;
  return fminf(fmaxf(q, -128.0f), 127.99609375f);
}
// quantize followed by hardtanh: [-1,1] subsumes the +-128 clamp -> single med3.
__device__ __forceinline__ float qhardtanh_q88(float x) {
  float q = rintf(x * 256.0f) * 0.00390625f;
  return fminf(fmaxf(q, -1.0f), 1.0f);
}

// ---------------------------------------------------------------------------
// Tensor Data Mover: 2-D tile load Global->LDS (D# per CDNA5 ISA 08 §8.3/8.4).
// count=1, type=2(image), no gather/iterate/pad, workgroup_mask=0 (no cluster).
// OOB columns (x >= tensor_d0) are zero-filled by the TDM.
// This toolchain's builtin takes 6 args (g0, g1, g2, g3, extra, cpol).
// ---------------------------------------------------------------------------
__device__ __forceinline__ void tdm_load_2d(
    u32 lds_addr, const void* gptr, u32 dsz_code,  // 0=1B,1=2B,2=4B,3=8B
    u32 tensor_d0, u32 tensor_d1, u32 tile_d0, u32 tile_d1, u32 stride_d0)
{
  unsigned long long ga = (unsigned long long)gptr;
  u32x4 g0;
  g0[0] = 1u;                                          // count=1, user mode
  g0[1] = lds_addr;                                    // [63:32] lds_addr
  g0[2] = (u32)ga;                                     // [95:64] global_addr lo
  g0[3] = (u32)((ga >> 32) & 0x01ffffffu) | (2u << 30);// global_addr hi | type=2
  i32x8 g1;
  g1[0] = (int)(dsz_code << 16);                       // mask=0 | data_size
  g1[1] = (int)((tensor_d0 & 0xffffu) << 16);          // [63:48] tensor_dim0 lo16
  g1[2] = (int)((tensor_d0 >> 16) | ((tensor_d1 & 0xffffu) << 16));
  g1[3] = (int)((tensor_d1 >> 16) | (tile_d0 << 16));  // [127:112] tile_dim0
  g1[4] = (int)(tile_d1 & 0xffffu);                    // tile_dim1 | tile_dim2=0
  g1[5] = (int)stride_d0;                              // tensor_dim0_stride lo32
  g1[6] = 0;                                           // stride hi16 | dim1_stride lo16
  g1[7] = 0;
  i32x4 gz4 = {0, 0, 0, 0};
  i32x8 gz8 = {0, 0, 0, 0, 0, 0, 0, 0};
  __builtin_amdgcn_tensor_load_to_lds(g0, g1, gz4, gz4, gz8, 0);
}

// ---------------------------------------------------------------------------
// Prep: quantize + f16-cast + pad all weights/biases into workspace.
// ---------------------------------------------------------------------------
__global__ void prep_kernel(
    const float* __restrict__ w1, const float* __restrict__ b1,
    const float* __restrict__ w2, const float* __restrict__ b2,
    const float* __restrict__ w3, const float* __restrict__ b3,
    const float* __restrict__ w4, const float* __restrict__ b4,
    const float* __restrict__ w5, const float* __restrict__ b5,
    const float* __restrict__ wl, const float* __restrict__ bl,
    char* __restrict__ ws)
{
  long id = (long)blockIdx.x * blockDim.x + threadIdx.x;

  // Transposed + K-padded linear weight: wlt[k][n] = quant(wl[n][k]); zeros in pad.
  if (id < (long)KPAD_L * 256) {
    int k = (int)(id >> 8);
    int n = (int)(id & 255);
    float v = (k < LFINAL) ? quantize_q88(wl[(long)n * LFINAL + k]) : 0.0f;
    ((_Float16*)(ws + OFF_WLT))[id] = (_Float16)v;
  }

  _Float16* wq = (_Float16*)(ws + OFF_W);
  if (id < 512) {                        // conv1: [16][32], KTOT = 3*8 = 24 (zero pad)
    int cout = (int)(id >> 5), kk = (int)(id & 31);
    float v = (kk < 24) ? quantize_q88(w1[cout * 24 + kk]) : 0.0f;
    wq[id] = (_Float16)v;
  } else if (id < 16896) {               // conv2..5: 4 x [16][256]; Cout zero-padded
    int j = (int)id - 512;
    int layer = j >> 12;                 // 0..3 -> w2,w3,w4,w5 (each KTOT = 256)
    int cout  = (j >> 8) & 15;
    int kk    = j & 255;
    const float* src = (layer == 0) ? w2 : (layer == 1) ? w3 : (layer == 2) ? w4 : w5;
    int cmax = 8 >> layer;               // 8,4,2,1
    float v = (cout < cmax) ? quantize_q88(src[cout * 256 + kk]) : 0.0f;
    wq[id] = (_Float16)v;
  }

  if (id >= 16896 && id < 16976) {       // conv biases, padded to 16 each, f32
    int j = (int)id - 16896;
    int layer = j >> 4, c = j & 15;
    const float* src = (layer == 0) ? b1 : (layer == 1) ? b2 :
                       (layer == 2) ? b3 : (layer == 3) ? b4 : b5;
    int cmax = 16 >> layer;              // 16,8,4,2,1
    ((float*)(ws + OFF_BIAS))[j] = (c < cmax) ? quantize_q88(src[c]) : 0.0f;
  }
  if (id >= 16976 && id < 17232) {       // linear bias
    int j = (int)id - 16976;
    ((float*)(ws + OFF_BL))[j] = quantize_q88(bl[j]);
  }
}

// ---------------------------------------------------------------------------
// One conv layer as a WMMA GEMM: M = Cout (padded 16), N = 16 positions,
// K = Cin*Ktaps in NCH chunks of 32.  Fragment layouts per CDNA5 ISA 7.12.2.
// ---------------------------------------------------------------------------
template<int CIN, int KT, int COUT, int NCH>
__device__ __forceinline__ void conv_stage_wmma(
    const _Float16* __restrict__ in, int istr,       // in[cin][pos], row stride istr
    const _Float16* __restrict__ w,                  // [16][NCH*32] padded weights
    const float* __restrict__ bias,                  // [16] padded
    _Float16* __restrict__ out, int ostr,            // out[cout][pos]
    int nout_r, int nvalid, int lane, int wave)
{
  const int KTOT = CIN * KT;
  // A fragment (16x32 f16): lane holds M = lane&15; low lanes K 0-7/16-23, high +8.
  const int m  = lane & 15;
  const int kh = (lane < 16) ? 0 : 8;
  const int mb = (lane < 16) ? 0 : 8;
  v16h a[NCH];
#pragma unroll
  for (int c = 0; c < NCH; ++c) {
    const _Float16* wr = w + m * (NCH * 32) + c * 32 + kh;
    v8h lo = *(const v8h*)(wr);          // 16B aligned
    v8h hi = *(const v8h*)(wr + 16);
#pragma unroll
    for (int e = 0; e < 8; ++e) { a[c][e] = lo[e]; a[c][8 + e] = hi[e]; }
  }
  // Hoist per-lane bias rows into registers (loop-invariant; avoids per-group DS).
  float bv[8];
#pragma unroll
  for (int r = 0; r < 8; ++r) bv[r] = bias[mb + r];

  const int ngroups = nout_r >> 4;
  for (int g = wave; g < ngroups; g += 8) {          // 8 waves stride over N-groups
    const int pos0 = g << 4;
    v8f acc = {0.f, 0.f, 0.f, 0.f, 0.f, 0.f, 0.f, 0.f};
#pragma unroll
    for (int c = 0; c < NCH; ++c) {
      const int n = lane & 15;                       // B column: lane&15 == N
      v16h bf;
#pragma unroll
      for (int s = 0; s < 16; ++s) {
        const int k = c * 32 + kh + (s < 8 ? s : s + 8);
        bf[s] = (k < KTOT) ? in[(k / KT) * istr + pos0 + n + (k % KT)] : (_Float16)0.0f;
      }
      acc = __builtin_amdgcn_wmma_f32_16x16x32_f16(false, a[c], false, bf,
                                                   (short)0, acc, false, false);
    }
    // D (16x16 f32): VGPR r -> M = r (+8 for high lanes), N = lane&15.
    const int col = pos0 + (lane & 15);
    if (col < nvalid) {
#pragma unroll
      for (int r = 0; r < 8; ++r) {
        const int cout = mb + r;
        if (cout < COUT)
          out[cout * ostr + col] = (_Float16)qhardtanh_q88(acc[r] + bv[r]);
      }
    }
  }
}

// ---------------------------------------------------------------------------
// Fused conv1..conv5 pipeline. One workgroup = (batch row, 512-wide tile).
// Weights + raw x window staged into LDS by the Tensor Data Mover (async DMA);
// all intermediates stay in LDS as f16 (~90 KB of the 320 KB WGP LDS).
// ---------------------------------------------------------------------------
__global__ __launch_bounds__(256) void conv_pipeline_kernel(
    const float* __restrict__ x, const char* __restrict__ ws,
    _Float16* __restrict__ h5)
{
  __shared__ __align__(16) char smem[S_TOTAL];
  _Float16* s_x  = (_Float16*)(smem + S_X_OFF);
  _Float16* s_h1 = (_Float16*)(smem + S_H1_OFF);
  _Float16* s_h2 = (_Float16*)(smem + S_H2_OFF);
  _Float16* s_h3 = (_Float16*)(smem + S_H3_OFF);
  _Float16* s_h4 = (_Float16*)(smem + S_H4_OFF);
  _Float16* s_w  = (_Float16*)(smem + S_W_OFF);
  float*    s_bias = (float*)(smem + S_BIAS_OFF);
  float*    s_xstg = (float*)(smem + S_XSTG_OFF);

  const int b = blockIdx.x;        // batch row
  const int t = blockIdx.y;        // tile
  const int tid = threadIdx.x;
  const int lane = tid & 31;
  const int wave = tid >> 5;

  const int base = t * TILE;
  const int n5 = (LFINAL - base < TILE) ? (LFINAL - base) : TILE;
  const int n4 = n5 + 127, n3 = n4 + 63, n2 = n3 + 31, n1 = n2 + 15, nx = n1 + 7;
  const int n5r = (n5 + 15) & ~15, n4r = (n4 + 15) & ~15, n3r = (n3 + 15) & ~15,
            n2r = (n2 + 15) & ~15, n1r = (n1 + 15) & ~15, nxr = (nx + 15) & ~15;

  // TDM: (1) weights+biases, one contiguous 34112B block as 4264 x 8B elements;
  //      (2) raw x window, 2-D tile [3 rows x nxr f32], row stride SEQ;
  //          tensor_d0 = SEQ-base so the TDM zero-fills past the row end.
  if (tid == 0) {
    tdm_load_2d((u32)(uintptr_t)(smem + S_W_OFF), ws + OFF_W,
                /*dsz=8B*/3u, 4264u, 1u, 4264u, 1u, 4264u);
    tdm_load_2d((u32)(uintptr_t)(smem + S_XSTG_OFF),
                x + (long)b * 3 * SEQ + base,
                /*dsz=4B*/2u, (u32)(SEQ - base), 3u, (u32)nxr, 3u, (u32)SEQ);
    __builtin_amdgcn_s_wait_tensorcnt((short)0);
  }
  __syncthreads();

  // quantize staged x window: f32 -> Q8.8 -> f16 (exact for |x|<8)
  for (int i = tid; i < 3 * nxr; i += 256) {
    int c = i / nxr, p = i - c * nxr;
    s_x[c * 784 + p] = (_Float16)quantize_q88(s_xstg[i]);
  }
  __syncthreads();

  conv_stage_wmma<3, 8, 16, 1>(s_x, 784, s_w,          s_bias,      s_h1, 752, n1r, n1r, lane, wave);
  __syncthreads();
  conv_stage_wmma<16, 16, 8, 8>(s_h1, 752, s_w + 512,  s_bias + 16, s_h2, 736, n2r, n2r, lane, wave);
  __syncthreads();
  conv_stage_wmma<8, 32, 4, 8>(s_h2, 736, s_w + 4608,  s_bias + 32, s_h3, 704, n3r, n3r, lane, wave);
  __syncthreads();
  conv_stage_wmma<4, 64, 2, 8>(s_h3, 704, s_w + 8704,  s_bias + 48, s_h4, 640, n4r, n4r, lane, wave);
  __syncthreads();
  // conv5 writes straight to global (post qhardtanh, f16), guarded to valid cols
  _Float16* h5row = h5 + (long)b * KPAD_L + base;
  conv_stage_wmma<2, 128, 1, 8>(s_h4, 640, s_w + 12800, s_bias + 64, h5row, 0, n5r, n5, lane, wave);

  if (t == NTILES - 1) {           // zero the K padding tail of this feature row
    for (int i = LFINAL + tid; i < KPAD_L; i += 256)
      h5[(long)b * KPAD_L + i] = (_Float16)0.0f;
  }
}

// ---------------------------------------------------------------------------
// Final linear: out[512][256] = quant(h5 @ wl^T + bl), f16 WMMA GEMM.
// One 16x16 output tile per wave; K = 7968 in chunks of 32.
// ---------------------------------------------------------------------------
__global__ __launch_bounds__(256) void linear_wmma_kernel(
    const _Float16* __restrict__ h5, const char* __restrict__ ws,
    float* __restrict__ out)
{
  const _Float16* wlt = (const _Float16*)(ws + OFF_WLT);   // [KPAD_L][256]
  const float* bl = (const float*)(ws + OFF_BL);
  const int tid = threadIdx.x, lane = tid & 31, wave = tid >> 5;
  const int tile = blockIdx.x * 8 + wave;   // 32 M-tiles x 16 N-tiles = 512 tiles
  const int m0 = (tile >> 4) << 4;
  const int n0 = (tile & 15) << 4;
  const int m  = lane & 15;
  const int kh = (lane < 16) ? 0 : 8;

  const _Float16* arow = h5 + (long)(m0 + m) * KPAD_L + kh;
  v8f acc = {0.f, 0.f, 0.f, 0.f, 0.f, 0.f, 0.f, 0.f};
#pragma unroll 2
  for (int kb = 0; kb < KPAD_L; kb += 32) {
    __builtin_prefetch(arow + kb + 256, 0, 1);             // global_prefetch next A
    v8h alo = *(const v8h*)(arow + kb);                    // 16B-aligned b128 loads
    v8h ahi = *(const v8h*)(arow + kb + 16);
    v16h a;
#pragma unroll
    for (int e = 0; e < 8; ++e) { a[e] = alo[e]; a[8 + e] = ahi[e]; }
    const _Float16* br = wlt + (long)(kb + kh) * 256 + n0 + (lane & 15); // lane&15 == N col
    v16h bf;
#pragma unroll
    for (int s = 0; s < 16; ++s) bf[s] = br[(long)(s < 8 ? s : s + 8) * 256];
    acc = __builtin_amdgcn_wmma_f32_16x16x32_f16(false, a, false, bf,
                                                 (short)0, acc, false, false);
  }
  const int n = lane & 15, mb = (lane < 16) ? 0 : 8;
  const float bias = bl[n0 + n];
#pragma unroll
  for (int r = 0; r < 8; ++r)
    out[(long)(m0 + mb + r) * 256 + n0 + n] = quantize_q88(acc[r] + bias);
}

// ---------------------------------------------------------------------------
extern "C" void kernel_launch(void* const* d_in, const int* in_sizes, int n_in,
                              void* d_out, int out_size, void* d_ws, size_t ws_size,
                              hipStream_t stream) {
  (void)in_sizes; (void)n_in; (void)out_size; (void)ws_size;
  const float* x  = (const float*)d_in[0];
  const float* w1 = (const float*)d_in[1];  const float* b1 = (const float*)d_in[2];
  const float* w2 = (const float*)d_in[3];  const float* b2 = (const float*)d_in[4];
  const float* w3 = (const float*)d_in[5];  const float* b3 = (const float*)d_in[6];
  const float* w4 = (const float*)d_in[7];  const float* b4 = (const float*)d_in[8];
  const float* w5 = (const float*)d_in[9];  const float* b5 = (const float*)d_in[10];
  const float* wl = (const float*)d_in[11]; const float* bl = (const float*)d_in[12];
  char* ws = (char*)d_ws;
  _Float16* h5 = (_Float16*)(ws + OFF_H5);

  prep_kernel<<<KPAD_L, 256, 0, stream>>>(w1, b1, w2, b2, w3, b3, w4, b4, w5, b5,
                                          wl, bl, ws);
  dim3 cgrid(512, NTILES);
  conv_pipeline_kernel<<<cgrid, 256, 0, stream>>>(x, ws, h5);
  linear_wmma_kernel<<<64, 256, 0, stream>>>(h5, ws, (float*)d_out);
}
